// VanillaAttention_21964462752136
// MI455X (gfx1250) — compile-verified, matrix operand order checked by reading, not run on hardware
//
#include <hip/hip_runtime.h>
#include <math.h>

#define BB  2
#define SS  2048
#define HH  1024
#define NHH 16
#define HDD 64
#define BSN (BB*SS)

typedef __attribute__((ext_vector_type(16))) __bf16         v16bf;
typedef __attribute__((ext_vector_type(8)))  __bf16         v8bf;
typedef __attribute__((ext_vector_type(8)))  float          v8f;
typedef __attribute__((ext_vector_type(4)))  float          f4;
typedef __attribute__((ext_vector_type(8)))  unsigned short us8;
typedef __attribute__((ext_vector_type(4)))  unsigned short us4;

static __device__ __forceinline__ void sched_fence() {
#if __has_builtin(__builtin_amdgcn_sched_barrier)
  __builtin_amdgcn_sched_barrier(0);   // pin pipeline phase boundaries
#endif
}

// f32 -> bf16, round-to-nearest-even (verified codegen: bfe + add3 + mov_b16).
static __device__ __forceinline__ unsigned short f2bf(float f) {
  unsigned u = __builtin_bit_cast(unsigned, f);
  u += 0x7FFFu + ((u >> 16) & 1u);
  return (unsigned short)(u >> 16);
}

// Pack 4 f32 -> 4 bf16, element-wise (this form keeps the b128 vector load;
// integer-pair packing variants provoked bad load scalarization).
static __device__ __forceinline__ us4 pk4(f4 v) {
  us4 r;
#pragma unroll
  for (int i = 0; i < 4; ++i) r[i] = f2bf(v[i]);
  return r;
}

// Async 16-byte global -> LDS copy (per-lane), tracked by ASYNCcnt.
static __device__ __forceinline__ void async_copy_b128(unsigned short* ldsDst,
                                                       const unsigned short* gsrc) {
  unsigned lds_off = (unsigned)(size_t)(const void*)ldsDst;  // low 32 bits = LDS addr
  asm volatile("global_load_async_to_lds_b128 %0, %1, off"
               :: "v"(lds_off), "v"(gsrc) : "memory");
}
static __device__ __forceinline__ void wait_async0() {
  asm volatile("s_wait_asynccnt 0x0" ::: "memory");
}

// Load one 16-lane-striped bf16 fragment (A or B operand) for
// v_wmma_f32_16x16x32_bf16. rowptr = row (lane&15) of a row-major [16][stride]
// bf16 tile at the k-block base. Lane half h holds k = h*8..+7 and 16+h*8..+7.
static __device__ __forceinline__ v16bf load_frag(const unsigned short* rowptr, int half) {
  us8 c0 = *(const us8*)(rowptr + half * 8);
  us8 c1 = *(const us8*)(rowptr + 16 + half * 8);
  v8bf b0 = __builtin_bit_cast(v8bf, c0);
  v8bf b1 = __builtin_bit_cast(v8bf, c1);
  v16bf r;
#pragma unroll
  for (int i = 0; i < 8; ++i) { r[i] = b0[i]; r[i + 8] = b1[i]; }
  return r;
}

static __device__ __forceinline__ v8f wmma_bf16(v16bf a, v16bf b, v8f c) {
  return __builtin_amdgcn_wmma_f32_16x16x32_bf16(false, a, false, b, (short)0, c,
                                                 false, false);
}

// ---------------------------------------------------------------------------
// Kernel 1: Y = x @ W^T + b for W in {Wq,Wk,Wv} (blockIdx.z selects), RoPE on
// Q/K, bf16 outputs. Q,K -> [B,NH,S,HD]; V -> [B,NH,HD,S] (transposed).
// 128x64 block tile, 8 waves, K-step 32, double-buffered + SW pipelined with
// sched_barrier fences: loads | WMMA | convert+store.
// ---------------------------------------------------------------------------
__global__ __launch_bounds__(256) void qkv_rope_kernel(
    const float* __restrict__ x,
    const float* __restrict__ Wq, const float* __restrict__ bq,
    const float* __restrict__ Wk, const float* __restrict__ bk,
    const float* __restrict__ Wv, const float* __restrict__ bv,
    const float* __restrict__ cosb, const float* __restrict__ sinb,
    unsigned short* __restrict__ qo, unsigned short* __restrict__ ko,
    unsigned short* __restrict__ vo) {
  const int tid = threadIdx.x, lane = tid & 31, wave = tid >> 5;
  const int half = (lane >> 4) & 1, col16 = lane & 15;
  const int mode = blockIdx.z;            // 0=Q 1=K 2=V
  const int row0 = blockIdx.x * 128;      // token block
  const int head = blockIdx.y;            // 64-wide N block == head
  const int n0 = head * HDD;

  const float* W    = (mode == 0) ? Wq : (mode == 1) ? Wk : Wv;
  const float* bias = (mode == 0) ? bq : (mode == 1) ? bk : bv;

  __shared__ unsigned short sA[2][128 * 32];  // x tile [m][k] bf16
  __shared__ unsigned short sB[2][64 * 32];   // W tile [n][k] bf16

  f4 rx[4], rw[2];                            // register staging (SW pipeline)
  auto gload = [&](int kt) {
#pragma unroll
    for (int i = 0; i < 4; ++i) {
      int c = tid + i * 256, m = c >> 3, k4 = (c & 7) * 4;
      rx[i] = *(const f4*)&x[(size_t)(row0 + m) * HH + kt * 32 + k4];
    }
#pragma unroll
    for (int i = 0; i < 2; ++i) {
      int c = tid + i * 256, n = c >> 3, k4 = (c & 7) * 4;
      rw[i] = *(const f4*)&W[(size_t)(n0 + n) * HH + kt * 32 + k4];
    }
  };
  auto lstore = [&](int buf) {
#pragma unroll
    for (int i = 0; i < 4; ++i) {
      int c = tid + i * 256, m = c >> 3, k4 = (c & 7) * 4;
      *(us4*)&sA[buf][m * 32 + k4] = pk4(rx[i]);
    }
#pragma unroll
    for (int i = 0; i < 2; ++i) {
      int c = tid + i * 256, n = c >> 3, k4 = (c & 7) * 4;
      *(us4*)&sB[buf][n * 32 + k4] = pk4(rw[i]);
    }
  };

  v8f acc[4] = {};

  gload(0);
  lstore(0);
  __syncthreads();
  for (int kt = 0; kt < HH / 32; ++kt) {
    const int cur = kt & 1;
    if (kt + 1 < HH / 32) gload(kt + 1);     // phase 1: issue global loads
    sched_fence();
    v16bf a = load_frag(&sA[cur][(wave * 16 + col16) * 32], half);
#pragma unroll
    for (int j = 0; j < 4; ++j) {            // phase 2: WMMA on current tile
      v16bf b = load_frag(&sB[cur][(j * 16 + col16) * 32], half);
      acc[j] = wmma_bf16(a, b, acc[j]);
    }
    sched_fence();
    if (kt + 1 < HH / 32) lstore(cur ^ 1);   // phase 3: convert + LDS store
    __syncthreads();
  }

  // bias
#pragma unroll
  for (int j = 0; j < 4; ++j) {
    float bb = bias[n0 + j * 16 + col16];
#pragma unroll
    for (int e = 0; e < 8; ++e) acc[j][e] += bb;
  }

  // RoPE: column d pairs with d+32 -> accumulator j pairs with j+2.
  if (mode < 2) {
#pragma unroll
    for (int j = 0; j < 2; ++j) {
      int d0 = j * 16 + col16, d1 = d0 + 32;
#pragma unroll
      for (int e = 0; e < 8; ++e) {
        int tok = row0 + wave * 16 + e + half * 8;
        int s = tok & (SS - 1);
        float c0 = cosb[s * HDD + d0], s0 = sinb[s * HDD + d0];
        float c1 = cosb[s * HDD + d1], s1 = sinb[s * HDD + d1];
        float q0 = acc[j][e], q1 = acc[j + 2][e];
        acc[j][e]     = q0 * c0 - q1 * s0;   // first half:  q*cos - q_hi*sin
        acc[j + 2][e] = q1 * c1 + q0 * s1;   // second half: q*cos + q_lo*sin
      }
    }
  }

#pragma unroll
  for (int j = 0; j < 4; ++j) {
    int d = j * 16 + col16;
#pragma unroll
    for (int e = 0; e < 8; ++e) {
      int tok = row0 + wave * 16 + e + half * 8;
      int bidx = tok >> 11;        // /S
      int s = tok & (SS - 1);
      unsigned short val = f2bf(acc[j][e]);
      if (mode == 2) {             // V transposed: [B,NH,HD,S]
        vo[(((size_t)bidx * NHH + head) * HDD + d) * SS + s] = val;
      } else {                     // Q/K: [B,NH,S,HD]
        unsigned short* dst = (mode == 0) ? qo : ko;
        dst[(((size_t)bidx * NHH + head) * SS + s) * HDD + d] = val;
      }
    }
  }
}

// ---------------------------------------------------------------------------
// Kernel 2: flash attention. Block = (b, h, 128 queries), 8 waves x 16 rows.
// Double-buffered K/V tiles staged by async global->LDS b128 copies: block
// kb+1 streams in (ASYNCcnt) while QK^T/softmax/PV run on block kb.
// Softmax runs in the exp2 domain (v_exp_f32 is natively 2^x).
// ---------------------------------------------------------------------------
__global__ __launch_bounds__(256) void attn_kernel(
    const unsigned short* __restrict__ q, const unsigned short* __restrict__ k,
    const unsigned short* __restrict__ vT, const int* __restrict__ mask,
    unsigned short* __restrict__ ctx) {
  const int tid = threadIdx.x, lane = tid & 31, wave = tid >> 5;
  const int half = (lane >> 4) & 1, col16 = lane & 15;
  const int qb = blockIdx.x, h = blockIdx.y, b = blockIdx.z;
  const size_t bh = (size_t)b * NHH + h;

  __shared__ unsigned short sK[2][64 * 64];     // [key][dim]
  __shared__ unsigned short sV[2][64 * 64];     // [dim][key]
  __shared__ unsigned short sP[8 * 16 * 64];    // wave-private P tiles
  __shared__ float sMask[2][64];

  auto issue_kv = [&](int kb2, int buf) {
    const int key0 = kb2 * 64;
    const unsigned short* gk = k + (bh * SS + key0) * HDD;   // 8KB contiguous
#pragma unroll
    for (int i = 0; i < 2; ++i) {
      int c = tid + i * 256;                                 // 512 x 16B chunks
      async_copy_b128(&sK[buf][c * 8], gk + c * 8);
    }
#pragma unroll
    for (int i = 0; i < 2; ++i) {
      int c = tid + i * 256;
      int d = c >> 3, off = (c & 7) * 8;                     // 64 rows x 128B
      async_copy_b128(&sV[buf][d * 64 + off],
                      vT + (bh * HDD + d) * SS + key0 + off);
    }
    if (tid < 64)
      sMask[buf][tid] = (mask[b * SS + key0 + tid] == 0) ? -1e30f : 0.0f;
  };

  // Q fragments (held in registers for the whole key loop)
  const unsigned short* qbase = q + (bh * SS + (size_t)qb * 128 + wave * 16) * HDD;
  v16bf aq[2];
  {
    const unsigned short* rp = qbase + (size_t)col16 * HDD;
    aq[0] = load_frag(rp, half);
    aq[1] = load_frag(rp + 32, half);
  }

  v8f o[4] = {};
  float mi[8], li[8];
#pragma unroll
  for (int e = 0; e < 8; ++e) { mi[e] = -1e30f; li[e] = 0.0f; }

  const float scale2 = 0.125f * 1.44269504088896340736f;  // (1/sqrt(64))*log2(e)

  issue_kv(0, 0);
  wait_async0();
  __syncthreads();

  for (int kb = 0; kb < SS / 64; ++kb) {
    const int cur = kb & 1;
    if (kb + 1 < SS / 64) issue_kv(kb + 1, cur ^ 1);  // async: overlaps compute

    // S = Q @ K^T (16 x 64 per wave)
    v8f sc[4] = {};
#pragma unroll
    for (int t = 0; t < 2; ++t)
#pragma unroll
      for (int j = 0; j < 4; ++j) {
        v16bf bk_ = load_frag(&sK[cur][(j * 16 + col16) * 64 + t * 32], half);
        sc[j] = wmma_bf16(aq[t], bk_, sc[j]);
      }

    // scale into exp2 domain + mask addend
#pragma unroll
    for (int j = 0; j < 4; ++j) {
      float madd = sMask[cur][j * 16 + col16];
#pragma unroll
      for (int e = 0; e < 8; ++e) sc[j][e] = sc[j][e] * scale2 + madd;
    }

    // online softmax; row stats reduced across the 16-lane half (wave32)
#pragma unroll
    for (int e = 0; e < 8; ++e) {
      float r = fmaxf(fmaxf(sc[0][e], sc[1][e]), fmaxf(sc[2][e], sc[3][e]));
#pragma unroll
      for (int m = 1; m < 16; m <<= 1) r = fmaxf(r, __shfl_xor(r, m, 32));
      float mnew = fmaxf(mi[e], r);
      float corr = exp2f(mi[e] - mnew);
      li[e] *= corr;
#pragma unroll
      for (int j = 0; j < 4; ++j) o[j][e] *= corr;
      float rs = 0.0f;
#pragma unroll
      for (int j = 0; j < 4; ++j) {
        float p = exp2f(sc[j][e] - mnew);
        sc[j][e] = p;
        rs += p;
      }
#pragma unroll
      for (int m = 1; m < 16; m <<= 1) rs += __shfl_xor(rs, m, 32);
      li[e] += rs;
      mi[e] = mnew;
    }

    // P (C layout) -> wave-private LDS in [row][key] so it reloads as A frags.
    unsigned short* pw = sP + wave * 16 * 64;
#pragma unroll
    for (int j = 0; j < 4; ++j)
#pragma unroll
      for (int e = 0; e < 8; ++e)
        pw[(e + half * 8) * 64 + j * 16 + col16] = f2bf(sc[j][e]);

    // O += P @ V  (same-wave LDS ops are in-order; no barrier needed)
#pragma unroll
    for (int t = 0; t < 2; ++t) {
      v16bf ap = load_frag(pw + col16 * 64 + t * 32, half);
#pragma unroll
      for (int j = 0; j < 4; ++j) {
        v16bf bv_ = load_frag(&sV[cur][(j * 16 + col16) * 64 + t * 32], half);
        o[j] = wmma_bf16(ap, bv_, o[j]);
      }
    }

    wait_async0();    // next buffer resident before anyone reads it
    __syncthreads();
  }

  // normalize + write context [B,S,H] bf16
  const size_t tok0 = (size_t)b * SS + (size_t)qb * 128 + wave * 16;
#pragma unroll
  for (int j = 0; j < 4; ++j) {
    int d = j * 16 + col16;
#pragma unroll
    for (int e = 0; e < 8; ++e) {
      int r = e + half * 8;
      float val = o[j][e] / li[e];
      ctx[(tok0 + r) * HH + h * HDD + d] = f2bf(val);
    }
  }
}

// ---------------------------------------------------------------------------
// Kernel 3: out = ctx @ Wo^T + bo (f32 output). ctx tile async-staged (already
// bf16); weight tile register-pipelined with element-wise conversion.
// ---------------------------------------------------------------------------
__global__ __launch_bounds__(256) void out_proj_kernel(
    const unsigned short* __restrict__ ctx, const float* __restrict__ Wo,
    const float* __restrict__ bo, float* __restrict__ out) {
  const int tid = threadIdx.x, lane = tid & 31, wave = tid >> 5;
  const int half = (lane >> 4) & 1, col16 = lane & 15;
  const int row0 = blockIdx.x * 128;
  const int n0 = blockIdx.y * 64;

  __shared__ unsigned short sA[2][128 * 32];
  __shared__ unsigned short sB[2][64 * 32];

  auto issueA = [&](int kt, int buf) {    // 128 rows x 64B -> 512 async chunks
#pragma unroll
    for (int i = 0; i < 2; ++i) {
      int c = tid + i * 256, m = c >> 2, off = (c & 3) * 8;
      async_copy_b128(&sA[buf][m * 32 + off],
                      ctx + (size_t)(row0 + m) * HH + kt * 32 + off);
    }
  };

  f4 rw[2];
  auto gloadB = [&](int kt) {
#pragma unroll
    for (int i = 0; i < 2; ++i) {
      int c = tid + i * 256, n = c >> 3, k4 = (c & 7) * 4;
      rw[i] = *(const f4*)&Wo[(size_t)(n0 + n) * HH + kt * 32 + k4];
    }
  };
  auto storeB = [&](int buf) {
#pragma unroll
    for (int i = 0; i < 2; ++i) {
      int c = tid + i * 256, n = c >> 3, k4 = (c & 7) * 4;
      *(us4*)&sB[buf][n * 32 + k4] = pk4(rw[i]);
    }
  };

  v8f acc[4] = {};
  issueA(0, 0);
  gloadB(0);
  storeB(0);
  wait_async0();
  __syncthreads();

  for (int kt = 0; kt < HH / 32; ++kt) {
    const int cur = kt & 1;
    if (kt + 1 < HH / 32) { issueA(kt + 1, cur ^ 1); gloadB(kt + 1); }
    sched_fence();
    v16bf a = load_frag(&sA[cur][(wave * 16 + col16) * 32], half);
#pragma unroll
    for (int j = 0; j < 4; ++j) {
      v16bf b = load_frag(&sB[cur][(j * 16 + col16) * 32], half);
      acc[j] = wmma_bf16(a, b, acc[j]);
    }
    sched_fence();
    if (kt + 1 < HH / 32) storeB(cur ^ 1);
    wait_async0();
    __syncthreads();
  }

#pragma unroll
  for (int j = 0; j < 4; ++j) {
    float bb = bo[n0 + j * 16 + col16];
#pragma unroll
    for (int e = 0; e < 8; ++e) {
      int tok = row0 + wave * 16 + e + half * 8;
      out[(size_t)tok * HH + n0 + j * 16 + col16] = acc[j][e] + bb;
    }
  }
}

extern "C" void kernel_launch(void* const* d_in, const int* in_sizes, int n_in,
                              void* d_out, int out_size, void* d_ws, size_t ws_size,
                              hipStream_t stream) {
  const float* x    = (const float*)d_in[0];
  const int*   mask = (const int*)d_in[1];
  const float* cosb = (const float*)d_in[2];
  const float* sinb = (const float*)d_in[3];
  const float* Wq = (const float*)d_in[4];  const float* bq = (const float*)d_in[5];
  const float* Wk = (const float*)d_in[6];  const float* bk = (const float*)d_in[7];
  const float* Wv = (const float*)d_in[8];  const float* bv = (const float*)d_in[9];
  const float* Wo = (const float*)d_in[10]; const float* bo = (const float*)d_in[11];

  // Workspace: 4 bf16 buffers of B*S*H (8 MiB each): q, k, vT, ctx
  unsigned short* qw = (unsigned short*)d_ws;
  unsigned short* kw = qw + (size_t)BSN * HH;
  unsigned short* vw = kw + (size_t)BSN * HH;
  unsigned short* cw = vw + (size_t)BSN * HH;
  float* out = (float*)d_out;

  qkv_rope_kernel<<<dim3(BSN / 128, NHH, 3), 256, 0, stream>>>(
      x, Wq, bq, Wk, bk, Wv, bv, cosb, sinb, qw, kw, vw);
  attn_kernel<<<dim3(SS / 128, NHH, BB), 256, 0, stream>>>(qw, kw, vw, mask, cw);
  out_proj_kernel<<<dim3(BSN / 128, HH / 64), 256, 0, stream>>>(cw, Wo, bo, out);
}
